// FlashAttention_50929722196041
// MI455X (gfx1250) — compile-verified, matrix operand order checked by reading, not run on hardware
//
#include <hip/hip_runtime.h>
#include <hip/hip_bf16.h>

// ---------------------------------------------------------------------------
// Types
// ---------------------------------------------------------------------------
typedef __attribute__((ext_vector_type(16))) _Float16     v16h;
typedef __attribute__((ext_vector_type(8)))  float        v8f;
typedef __attribute__((ext_vector_type(4)))  unsigned int u32x4;
typedef __attribute__((ext_vector_type(4)))  float        f32x4;

union Frag16 {
    v16h     v;
    u32x4    u[2];
    _Float16 h[16];
};

__device__ __forceinline__ v8f wmma_f16(v16h a, v16h b, v8f c) {
    // emits v_wmma_f32_16x16x32_f16
    return __builtin_amdgcn_wmma_f32_16x16x32_f16(
        /*neg_a=*/false, a, /*neg_b=*/false, b,
        /*c_mod=*/(short)0, c, /*reuse_a=*/false, /*reuse_b=*/false);
}

#define HD    128
#define SEQ   2048
#define DIMM  2048
#define NH    16
#define NKV   4
#define GSZ   (NH / NKV)
#define KVW   (NKV * HD)                 // 512
#define ATT_SCALE 0.08838834764831845f   // 1/sqrt(128)

// ---------------------------------------------------------------------------
// GEMM: C[M,Nc] = A[M,Kc] * B[Kc,Nc]
//   A: f32 (A_HALF=false) or f16 (A_HALF=true); B: f32 -> f16 in LDS
//   C: f16 (OUT_HALF=true) or f32 (OUT_HALF=false)
//   Block: 256 threads, 128x128 tile, BK=32, wave grid 4x2 (32x64 per wave).
//   Register-staged double buffering: next tile's global loads overlap WMMAs.
// ---------------------------------------------------------------------------
template<bool A_HALF, bool OUT_HALF>
__global__ __launch_bounds__(256)
void gemm_f16_wmma(const void* __restrict__ Ap, const float* __restrict__ Bp,
                   void* __restrict__ Cp, int M, int Nc, int Kc)
{
    __shared__ _Float16 Al[128][32];  // [m][k]
    __shared__ _Float16 Bt[128][32];  // [n][k]  (B transposed)

    const int t    = threadIdx.x;
    const int lane = t & 31;
    const int w    = t >> 5;
    const int half = lane >> 4;
    const int l16  = lane & 15;

    const int n0 = blockIdx.x * 128;
    const int m0 = blockIdx.y * 128;
    const int wm = (w & 3) * 32;   // wave row base inside tile
    const int wn = (w >> 2) * 64;  // wave col base inside tile

    // loader thread mapping
    const int arow = t >> 1, akb = (t & 1) * 16;   // A: 2 thr/row, 16 k each
    const int bk   = t >> 3, bnb = (t & 7) * 16;   // B: 8 thr/krow, 16 n each

    // register staging for the next tile
    _Float16 ra[16];
    float    rb[16];

    auto loadA = [&](int k0) {
        if (A_HALF) {
            const u32x4* src = (const u32x4*)((const _Float16*)Ap +
                               (size_t)(m0 + arow) * Kc + k0 + akb);
            *(u32x4*)&ra[0] = src[0];
            *(u32x4*)&ra[8] = src[1];
        } else {
            const f32x4* src = (const f32x4*)((const float*)Ap +
                               (size_t)(m0 + arow) * Kc + k0 + akb);
            f32x4 f0 = src[0], f1 = src[1], f2 = src[2], f3 = src[3];
            ra[0]  = (_Float16)f0[0]; ra[1]  = (_Float16)f0[1];
            ra[2]  = (_Float16)f0[2]; ra[3]  = (_Float16)f0[3];
            ra[4]  = (_Float16)f1[0]; ra[5]  = (_Float16)f1[1];
            ra[6]  = (_Float16)f1[2]; ra[7]  = (_Float16)f1[3];
            ra[8]  = (_Float16)f2[0]; ra[9]  = (_Float16)f2[1];
            ra[10] = (_Float16)f2[2]; ra[11] = (_Float16)f2[3];
            ra[12] = (_Float16)f3[0]; ra[13] = (_Float16)f3[1];
            ra[14] = (_Float16)f3[2]; ra[15] = (_Float16)f3[3];
        }
    };
    auto loadB = [&](int k0) {
        const f32x4* src = (const f32x4*)(Bp + (size_t)(k0 + bk) * Nc + n0 + bnb);
        f32x4 f0 = src[0], f1 = src[1], f2 = src[2], f3 = src[3];
        rb[0]  = f0[0]; rb[1]  = f0[1]; rb[2]  = f0[2]; rb[3]  = f0[3];
        rb[4]  = f1[0]; rb[5]  = f1[1]; rb[6]  = f1[2]; rb[7]  = f1[3];
        rb[8]  = f2[0]; rb[9]  = f2[1]; rb[10] = f2[2]; rb[11] = f2[3];
        rb[12] = f3[0]; rb[13] = f3[1]; rb[14] = f3[2]; rb[15] = f3[3];
    };
    auto storeTiles = [&]() {
        *(u32x4*)&Al[arow][akb]     = *(u32x4*)&ra[0];
        *(u32x4*)&Al[arow][akb + 8] = *(u32x4*)&ra[8];
        for (int e = 0; e < 16; ++e)
            Bt[bnb + e][bk] = (_Float16)rb[e];
    };

    v8f acc[2][4];
    for (int i = 0; i < 2; ++i)
        for (int j = 0; j < 4; ++j)
            acc[i][j] = (v8f)0.0f;

    loadA(0);
    loadB(0);
    for (int k0 = 0; k0 < Kc; k0 += 32) {
        storeTiles();
        __syncthreads();

        // prefetch next tile while this tile computes (LOADcnt path)
        if (k0 + 32 < Kc) { loadA(k0 + 32); loadB(k0 + 32); }

        // fragments (batched ds_load_b128) + WMMA chain
        Frag16 af[2], bf[4];
        for (int mi = 0; mi < 2; ++mi) {
            const int row = wm + mi * 16 + l16;
            af[mi].u[0] = *(const u32x4*)&Al[row][half * 8];
            af[mi].u[1] = *(const u32x4*)&Al[row][16 + half * 8];
        }
        for (int ni = 0; ni < 4; ++ni) {
            const int col = wn + ni * 16 + l16;
            bf[ni].u[0] = *(const u32x4*)&Bt[col][half * 8];
            bf[ni].u[1] = *(const u32x4*)&Bt[col][16 + half * 8];
        }
        for (int mi = 0; mi < 2; ++mi)
            for (int ni = 0; ni < 4; ++ni)
                acc[mi][ni] = wmma_f16(af[mi].v, bf[ni].v, acc[mi][ni]);

        __syncthreads();
    }

    // epilogue: C layout row = vgpr + 8*(lane/16), col = lane%16
    for (int mi = 0; mi < 2; ++mi)
        for (int ni = 0; ni < 4; ++ni)
            for (int v = 0; v < 8; ++v) {
                const int row = m0 + wm + mi * 16 + v + 8 * half;
                const int col = n0 + wn + ni * 16 + l16;
                const float val = acc[mi][ni][v];
                if (OUT_HALF)
                    ((_Float16*)Cp)[(size_t)row * Nc + col] = (_Float16)val;
                else
                    ((float*)Cp)[(size_t)row * Nc + col] = val;
            }
}

// ---------------------------------------------------------------------------
// Flash attention (causal, GQA): one block per (batch, head, 128-query tile).
// 8 waves x 16 query rows. K tile 32x128 and V^T tile 128x32 shared in LDS.
// Register-staged double buffering of the K/V tiles.
// ---------------------------------------------------------------------------
__global__ __launch_bounds__(256)
void attn_kernel(const _Float16* __restrict__ Q, const _Float16* __restrict__ K,
                 const _Float16* __restrict__ V, _Float16* __restrict__ AO)
{
    __shared__ _Float16 Kl[32][128];     // [key][d]
    __shared__ _Float16 Vt[128][32];     // [d][key]
    __shared__ _Float16 Pl[8][16][32];   // per-wave P staging [row][key]

    const int t    = threadIdx.x;
    const int lane = t & 31;
    const int w    = t >> 5;
    const int half = lane >> 4;
    const int l16  = lane & 15;

    const int qt = blockIdx.x;          // query tile (0..15)
    const int h  = blockIdx.y;          // head (0..15)
    const int b  = blockIdx.z;          // batch (0..1)
    const int kh = h / GSZ;             // GQA: head h -> kv head h/4

    const int q0 = qt * 128 + w * 16;   // wave's query base

    // ---- load Q fragments (16 rows x 128 d), A-matrix layout ----
    Frag16 qf[4];
    {
        const u32x4* Qv = (const u32x4*)(Q + (size_t)(b * SEQ + q0 + l16) * DIMM +
                                         (size_t)h * HD);
        for (int c = 0; c < 4; ++c) {
            qf[c].u[0] = Qv[c * 4 + half];       // d = c*32 + half*8 .. +7
            qf[c].u[1] = Qv[c * 4 + 2 + half];   // d = c*32 + 16 + half*8 .. +7
        }
    }

    // K/V register staging
    u32x4    rk[2];
    _Float16 rv[16];
    const int vkey = t >> 3, vdb = (t & 7) * 16;   // V loader mapping

    auto loadKV = [&](int kt) {
        const _Float16* Kbase = K + (size_t)(b * SEQ + kt * 32) * KVW + (size_t)kh * HD;
        rk[0] = ((const u32x4*)(Kbase + (size_t)(t >> 4) * KVW))[t & 15];
        rk[1] = ((const u32x4*)(Kbase + (size_t)((t + 256) >> 4) * KVW))[(t + 256) & 15];
        const u32x4* vs = (const u32x4*)(V +
            (size_t)(b * SEQ + kt * 32 + vkey) * KVW + (size_t)kh * HD + vdb);
        *(u32x4*)&rv[0] = vs[0];
        *(u32x4*)&rv[8] = vs[1];
    };
    auto storeKV = [&]() {
        ((u32x4*)Kl)[t]       = rk[0];
        ((u32x4*)Kl)[t + 256] = rk[1];
        for (int e = 0; e < 16; ++e)
            Vt[vdb + e][vkey] = rv[e];
    };

    v8f o[8];
    for (int j = 0; j < 8; ++j) o[j] = (v8f)0.0f;
    float m8[8], l8[8];
    for (int v = 0; v < 8; ++v) { m8[v] = -1e30f; l8[v] = 0.0f; }

    const int nkt = qt * 4 + 4;  // causal: keys up to qt*128+127

    loadKV(0);
    for (int kt = 0; kt < nkt; ++kt) {
        storeKV();
        __syncthreads();

        // prefetch next K/V tile while this one computes
        if (kt + 1 < nkt) loadKV(kt + 1);

        // ---- S = Q K^T : two 16x16 subtiles, batched frags + 4 WMMAs ----
        v8f s[2];
        for (int st = 0; st < 2; ++st) {
            const int key = st * 16 + l16;   // B operand: column = key
            Frag16 kf[4];
            for (int c = 0; c < 4; ++c) {
                kf[c].u[0] = *(const u32x4*)&Kl[key][c * 32 + half * 8];
                kf[c].u[1] = *(const u32x4*)&Kl[key][c * 32 + 16 + half * 8];
            }
            s[st] = (v8f)0.0f;
            for (int c = 0; c < 4; ++c)
                s[st] = wmma_f16(qf[c].v, kf[c].v, s[st]);
        }

        // ---- scale, causal mask, online softmax stats ----
        float mnew[8];
        for (int v = 0; v < 8; ++v) {
            const int qg = q0 + v + 8 * half;
            float mx = -1e30f;
            for (int st = 0; st < 2; ++st) {
                const int kg = kt * 32 + st * 16 + l16;
                float sv = s[st][v] * ATT_SCALE;
                if (kg > qg) sv = -1e30f;
                s[st][v] = sv;
                mx = fmaxf(mx, sv);
            }
            for (int off = 1; off < 16; off <<= 1)
                mx = fmaxf(mx, __shfl_xor(mx, off, 32));
            mnew[v] = fmaxf(m8[v], mx);
        }
        for (int v = 0; v < 8; ++v) {
            float rs = 0.0f;
            for (int st = 0; st < 2; ++st) {
                const float p = __expf(s[st][v] - mnew[v]);
                s[st][v] = p;
                rs += p;
            }
            for (int off = 1; off < 16; off <<= 1)
                rs += __shfl_xor(rs, off, 32);
            const float sc = __expf(m8[v] - mnew[v]);
            l8[v] = l8[v] * sc + rs;
            m8[v] = mnew[v];
            for (int j = 0; j < 8; ++j)
                o[j][v] *= sc;
        }

        // ---- P: C-layout -> A-layout via per-wave LDS staging ----
        for (int v = 0; v < 8; ++v) {
            const int row = v + 8 * half;
            Pl[w][row][l16]      = (_Float16)s[0][v];
            Pl[w][row][16 + l16] = (_Float16)s[1][v];
        }
        Frag16 pf;
        pf.u[0] = *(const u32x4*)&Pl[w][l16][half * 8];
        pf.u[1] = *(const u32x4*)&Pl[w][l16][16 + half * 8];

        // ---- O += P V : 8 column tiles, batched frags in groups of 4 ----
        for (int g = 0; g < 2; ++g) {
            Frag16 vf[4];
            for (int j = 0; j < 4; ++j) {
                const int d = (g * 4 + j) * 16 + l16;   // B operand: column = d
                vf[j].u[0] = *(const u32x4*)&Vt[d][half * 8];
                vf[j].u[1] = *(const u32x4*)&Vt[d][16 + half * 8];
            }
            for (int j = 0; j < 4; ++j)
                o[g * 4 + j] = wmma_f16(pf.v, vf[j].v, o[g * 4 + j]);
        }

        __syncthreads();
    }

    // ---- normalize and write attention output [b*SEQ+q, h*HD+d] as f16 ----
    for (int v = 0; v < 8; ++v) {
        const float inv = 1.0f / l8[v];
        const size_t row = (size_t)(b * SEQ + q0 + v + 8 * half);
        for (int j = 0; j < 8; ++j) {
            const int col = h * HD + j * 16 + l16;
            AO[row * DIMM + col] = (_Float16)(o[j][v] * inv);
        }
    }
}

// ---------------------------------------------------------------------------
// Host launcher
// ---------------------------------------------------------------------------
extern "C" void kernel_launch(void* const* d_in, const int* in_sizes, int n_in,
                              void* d_out, int out_size, void* d_ws, size_t ws_size,
                              hipStream_t stream)
{
    (void)in_sizes; (void)n_in; (void)out_size; (void)ws_size;

    const float* x  = (const float*)d_in[0];
    const float* Wq = (const float*)d_in[1];
    const float* Wk = (const float*)d_in[2];
    const float* Wv = (const float*)d_in[3];
    const float* Wo = (const float*)d_in[4];
    float* out = (float*)d_out;

    const size_t ROWS = 2 * (size_t)SEQ;  // 4096
    _Float16* Qb  = (_Float16*)d_ws;                   // 4096 x 2048
    _Float16* Kb  = Qb + ROWS * DIMM;                  // 4096 x 512
    _Float16* Vb  = Kb + ROWS * KVW;                   // 4096 x 512
    _Float16* AOb = Vb + ROWS * KVW;                   // 4096 x 2048

    const dim3 blk(256);

    // QKV projections (f32 in -> f16 out)
    gemm_f16_wmma<false, true><<<dim3(DIMM / 128, ROWS / 128), blk, 0, stream>>>(
        x, Wq, Qb, (int)ROWS, DIMM, DIMM);
    gemm_f16_wmma<false, true><<<dim3(KVW / 128, ROWS / 128), blk, 0, stream>>>(
        x, Wk, Kb, (int)ROWS, KVW, DIMM);
    gemm_f16_wmma<false, true><<<dim3(KVW / 128, ROWS / 128), blk, 0, stream>>>(
        x, Wv, Vb, (int)ROWS, KVW, DIMM);

    // Causal GQA flash attention
    attn_kernel<<<dim3(SEQ / 128, NH, 2), blk, 0, stream>>>(Qb, Kb, Vb, AOb);

    // Output projection (f16 A, f32 out)
    gemm_f16_wmma<true, false><<<dim3(DIMM / 128, ROWS / 128), blk, 0, stream>>>(
        AOb, Wo, out, (int)ROWS, DIMM, DIMM);
}
